// MoPNetwork_82987358093419
// MI455X (gfx1250) — matrix operand order repaired and checked by reading, not compiled, not run on hardware
//
#include <hip/hip_runtime.h>
#include <hip/hip_bf16.h>

typedef _Float16 f16;
typedef __attribute__((ext_vector_type(16))) _Float16 v16h;
typedef __attribute__((ext_vector_type(8)))  _Float16 v8h;
typedef __attribute__((ext_vector_type(8)))  float    v8f;
typedef __attribute__((ext_vector_type(4)))  unsigned int u32x4;
typedef __attribute__((ext_vector_type(8)))  int i32x8;
typedef __attribute__((ext_vector_type(4)))  int i32x4;

#define OBS_DIM 256
#define GOAL_DIM 64
#define NUM_SKILLS 16
#define ACTION_DIM 32
#define HIDDEN 128
#define BATCH 65536

// Workspace layout (f16 element offsets): weights transposed to [out][in]
#define WA1T_OFF 0                                        // [128][320]
#define WA2T_OFF (WA1T_OFF + HIDDEN*(OBS_DIM+GOAL_DIM))   // [16][128]
#define W1T_OFF  (WA2T_OFF + NUM_SKILLS*HIDDEN)           // [16][128][256]
#define W2T_OFF  (W1T_OFF + NUM_SKILLS*HIDDEN*OBS_DIM)    // [16][32][128]
#define WS_F16_TOTAL (W2T_OFF + NUM_SKILLS*ACTION_DIM*HIDDEN)

#define WAVES 4
#define TILES 2                                  // 16-row tiles per wave
#define ROWS_PER_WAVE (TILES*16)                 // 32
#define ROWS_PER_BLOCK (WAVES*ROWS_PER_WAVE)     // 128

#if __has_builtin(__builtin_amdgcn_tensor_load_to_lds)
#define HAVE_TDM 1
#define W1BUFS 2
#else
#define HAVE_TDM 0
#define W1BUFS 1
#endif

// ---------------------------------------------------------------------------
// Weight conversion: fp32 [in][out] -> f16 [out][in] (B-fragment friendly)
// ---------------------------------------------------------------------------
__global__ void convert_weights_kernel(const float* __restrict__ Wa1,
                                       const float* __restrict__ Wa2,
                                       const float* __restrict__ W1,
                                       const float* __restrict__ W2,
                                       f16* __restrict__ ws) {
  for (int i = blockIdx.x * blockDim.x + threadIdx.x; i < WS_F16_TOTAL;
       i += gridDim.x * blockDim.x) {
    float val;
    if (i < WA2T_OFF) {
      int idx = i - WA1T_OFF;
      int n = idx / (OBS_DIM + GOAL_DIM), k = idx % (OBS_DIM + GOAL_DIM);
      val = Wa1[k * HIDDEN + n];
    } else if (i < W1T_OFF) {
      int idx = i - WA2T_OFF;
      int n = idx / HIDDEN, k = idx % HIDDEN;
      val = Wa2[k * NUM_SKILLS + n];
    } else if (i < W2T_OFF) {
      int idx = i - W1T_OFF;
      int s = idx / (HIDDEN * OBS_DIM), r = idx % (HIDDEN * OBS_DIM);
      int n = r / OBS_DIM, k = r % OBS_DIM;
      val = W1[(s * OBS_DIM + k) * HIDDEN + n];
    } else {
      int idx = i - W2T_OFF;
      int s = idx / (ACTION_DIM * HIDDEN), r = idx % (ACTION_DIM * HIDDEN);
      int n = r / HIDDEN, k = r % HIDDEN;
      val = W2[(s * HIDDEN + k) * ACTION_DIM + n];
    }
    ws[i] = (f16)val;
  }
}

// ---------------------------------------------------------------------------
// Tensor Data Mover: DMA one 64 KB W1T[s] tile (8192 x 8-byte elements, 1-D)
// from global into LDS. Tracked by TENSORcnt (wait via s_wait_tensorcnt).
// D# per CDNA5 ISA ch.8: group0 {count=1, lds_addr, global_addr, type=2},
// group1 {data_size=3(8B), tensor_dim0=8192, tensor_dim1=1, tile_dim0=8192,
//         tile_dim1=1, tensor_dim0_stride=8192}.
// ---------------------------------------------------------------------------
#if HAVE_TDM
__device__ __forceinline__ void tdm_load_64k(const void* gsrc, void* lds_dst) {
  const uint64_t ga = (uint64_t)(uintptr_t)gsrc;
  const uint32_t la = (uint32_t)(uintptr_t)lds_dst;  // LDS byte offset (low 32b of aperture ptr)
  u32x4 g0;
  g0[0] = 1u;                                        // count = 1 (valid user descriptor)
  g0[1] = la;                                        // lds_addr
  g0[2] = (uint32_t)ga;                              // global_addr[31:0]
  g0[3] = (uint32_t)((ga >> 32) & 0x01FFFFFFu) | (2u << 30);  // global_addr[56:32] | type=2
  i32x8 g1;
  g1[0] = (int)(3u << 16);                           // workgroup_mask=0, data_size=3 (8B)
  g1[1] = (int)(0x2000u << 16);                      // tensor_dim0 = 8192 (low 16)
  g1[2] = (int)(1u << 16);                           // tensor_dim0 hi=0 | tensor_dim1 = 1
  g1[3] = (int)(0x2000u << 16);                      // tensor_dim1 hi=0 | tile_dim0 = 8192
  g1[4] = 1;                                         // tile_dim1 = 1, tile_dim2 = 0
  g1[5] = 0x2000;                                    // tensor_dim0_stride = 8192
  g1[6] = 0;
  g1[7] = 0;
  i32x4 z4 = {0, 0, 0, 0};
#if defined(__clang_major__) && (__clang_major__ >= 23)
  i32x8 z8 = {0, 0, 0, 0, 0, 0, 0, 0};
  __builtin_amdgcn_tensor_load_to_lds(g0, g1, z4, z4, z8, 0);
#else
  __builtin_amdgcn_tensor_load_to_lds(g0, g1, z4, z4, 0);
#endif
}
#endif

// ---------------------------------------------------------------------------
// WMMA fragment helpers (wave32, V_WMMA_F32_16X16X32_F16 layouts per CDNA5 ISA)
// ---------------------------------------------------------------------------
// A (16x32 tile of a row-major fp32 matrix in GLOBAL memory, converted f16):
// lane holds row r=lane&15; two contiguous 8-elem chunks at k = kbase+8*half
// and k = kbase+16+8*half -> 4x global_load_b128 + cvt.
__device__ __forceinline__ v16h load_a_gf32(const float* __restrict__ base, int ldk,
                                            int row0, int kbase, int lane) {
  const int r = lane & 15;
  const int h8 = (lane >> 4) << 3;
  const float* p = base + (size_t)(row0 + r) * ldk + kbase + h8;
  v16h out;
#pragma unroll
  for (int i = 0; i < 8; ++i) out[i] = (f16)p[i];
#pragma unroll
  for (int i = 0; i < 8; ++i) out[i + 8] = (f16)p[16 + i];
  return out;
}

// A from LDS (f16 row-major, used for the hidden activations)
__device__ __forceinline__ v16h load_a_lds(const f16* base, int ldk, int kbase, int lane) {
  const int r = lane & 15;
  const int h8 = (lane >> 4) << 3;
  const f16* p = base + r * ldk + kbase + h8;
  v8h a0 = *(const v8h*)p;
  v8h a1 = *(const v8h*)(p + 16);
  v16h out;
#pragma unroll
  for (int i = 0; i < 8; ++i) { out[i] = a0[i]; out[i + 8] = a1[i]; }
  return out;
}

// B (32x16, weights stored transposed [N][K] f16): lane holds column
// n = nbase + (lane&15); two 8-elem chunks at k = kbase + 8*half and kbase + 16 + 8*half.
// Works for both global and (after inlining) LDS pointers.
__device__ __forceinline__ v16h load_b(const f16* wt, int ldk, int nbase, int kbase, int lane) {
  const f16* p = wt + (size_t)(nbase + (lane & 15)) * ldk + kbase + ((lane >> 4) << 3);
  v8h b0 = *(const v8h*)p;
  v8h b1 = *(const v8h*)(p + 16);
  v16h out;
#pragma unroll
  for (int i = 0; i < 8; ++i) { out[i] = b0[i]; out[i + 8] = b1[i]; }
  return out;
}

__device__ __forceinline__ v8f wmma_f16(v16h a, v16h b, v8f c) {
  return __builtin_amdgcn_wmma_f32_16x16x32_f16(false, a, false, b, (short)0, c,
                                                false, false);
}

// ---------------------------------------------------------------------------
// Main fused kernel: gate MLP + softmax + 16 experts + alpha-weighted mix.
// 4 waves/block; each wave owns two 16-row tiles (32 rows). W1[s] tiles are
// double-buffered in LDS via the Tensor Data Mover, overlapped with compute.
// ---------------------------------------------------------------------------
__global__ __launch_bounds__(WAVES * 32) void mop_kernel(
    const float* __restrict__ obs, const float* __restrict__ goals,
    const float* __restrict__ ba1, const float* __restrict__ ba2,
    const float* __restrict__ b1, const float* __restrict__ b2,
    const f16* __restrict__ ws,
    float* __restrict__ out_mixed, float* __restrict__ out_alphas) {

  __shared__ __align__(16) f16 lds_w1[W1BUFS][HIDDEN * OBS_DIM];     // 64/128 KB
  __shared__ __align__(16) f16 lds_h[WAVES][ROWS_PER_WAVE * HIDDEN]; // 32 KB
  __shared__ __align__(16) float lds_sc[WAVES][ROWS_PER_WAVE * 16];  //  8 KB
  __shared__ __align__(16) float lds_al[WAVES][ROWS_PER_WAVE * 16];  //  8 KB

  const int lane = threadIdx.x & 31;
  const int w = threadIdx.x >> 5;
  const int half = lane >> 4;
  const int n16 = lane & 15;
  const int rowbase = blockIdx.x * ROWS_PER_BLOCK + w * ROWS_PER_WAVE;

  const f16* Wa1T = ws + WA1T_OFF;
  const f16* Wa2T = ws + WA2T_OFF;
  const f16* W1T  = ws + W1T_OFF;
  const f16* W2T  = ws + W2T_OFF;

#if HAVE_TDM
  // Kick off async DMA of the first two W1 tiles; they land while the whole
  // gate MLP computes (TENSORcnt tracks completion).
  if (w == 0) {
    tdm_load_64k(W1T, lds_w1[0]);
    tdm_load_64k(W1T + (size_t)HIDDEN * OBS_DIM, lds_w1[1]);
  }
#endif

  // ---- gate layer 1: ha = relu([obs|goals] @ Wa1 + ba1), both tiles ----
#pragma unroll 1
  for (int nt = 0; nt < HIDDEN / 16; ++nt) {
    v8f acc[TILES];
#pragma unroll
    for (int t = 0; t < TILES; ++t) acc[t] = (v8f){};
#pragma unroll
    for (int kk = 0; kk < OBS_DIM / 32; ++kk) {
      v16h b = load_b(Wa1T, OBS_DIM + GOAL_DIM, nt * 16, kk * 32, lane);
#pragma unroll
      for (int t = 0; t < TILES; ++t) {
        v16h a = load_a_gf32(obs, OBS_DIM, rowbase + t * 16, kk * 32, lane);
        acc[t] = wmma_f16(a, b, acc[t]);
      }
    }
#pragma unroll
    for (int kk = 0; kk < GOAL_DIM / 32; ++kk) {
      v16h b = load_b(Wa1T, OBS_DIM + GOAL_DIM, nt * 16, OBS_DIM + kk * 32, lane);
#pragma unroll
      for (int t = 0; t < TILES; ++t) {
        v16h a = load_a_gf32(goals, GOAL_DIM, rowbase + t * 16, kk * 32, lane);
        acc[t] = wmma_f16(a, b, acc[t]);
      }
    }
    float bias = ba1[nt * 16 + n16];
#pragma unroll
    for (int t = 0; t < TILES; ++t)
#pragma unroll
      for (int v = 0; v < 8; ++v) {
        float x = acc[t][v] + bias;
        x = x > 0.f ? x : 0.f;
        lds_h[w][(t * 16 + v + 8 * half) * HIDDEN + nt * 16 + n16] = (f16)x;
      }
  }

  // ---- gate layer 2: logits = ha @ Wa2 + ba2 ----
#pragma unroll
  for (int t = 0; t < TILES; ++t) {
    v8f acc = {};
#pragma unroll
    for (int kk = 0; kk < HIDDEN / 32; ++kk) {
      v16h a = load_a_lds(lds_h[w] + t * 16 * HIDDEN, HIDDEN, kk * 32, lane);
      v16h b = load_b(Wa2T, HIDDEN, 0, kk * 32, lane);
      acc = wmma_f16(a, b, acc);
    }
    float bias = ba2[n16];
#pragma unroll
    for (int v = 0; v < 8; ++v)
      lds_sc[w][(t * 16 + v + 8 * half) * 16 + n16] = acc[v] + bias;
  }
  __syncthreads();

  // ---- softmax over skills: one of the wave's 32 rows per lane ----
  {
    float vals[16];
    float m = -3.4e38f;
#pragma unroll
    for (int j = 0; j < 16; ++j) {
      vals[j] = lds_sc[w][lane * 16 + j];
      m = vals[j] > m ? vals[j] : m;
    }
    float sum = 0.f;
#pragma unroll
    for (int j = 0; j < 16; ++j) { float e = __expf(vals[j] - m); vals[j] = e; sum += e; }
    float inv = 1.f / sum;
    float* adst = out_alphas + (size_t)(rowbase + lane) * NUM_SKILLS;
#pragma unroll
    for (int j = 0; j < 16; ++j) {
      float aj = vals[j] * inv;
      lds_al[w][lane * 16 + j] = aj;
      adst[j] = aj;
    }
  }
  __syncthreads();

  // ---- experts ----
  v8f mix[TILES][2];
#pragma unroll
  for (int t = 0; t < TILES; ++t) { mix[t][0] = (v8f){}; mix[t][1] = (v8f){}; }

#pragma unroll 1
  for (int s = 0; s < NUM_SKILLS; ++s) {
#if HAVE_TDM
    // Wait until this skill's DMA has landed (keep s+1 in flight), then sync.
    if (w == 0) {
      if (s + 1 < NUM_SKILLS) __builtin_amdgcn_s_wait_tensorcnt(1);
      else                    __builtin_amdgcn_s_wait_tensorcnt(0);
    }
    __syncthreads();
    const f16* w1buf = lds_w1[s & 1];
#else
    __syncthreads();
    {
      const int4* src = (const int4*)(W1T + (size_t)s * HIDDEN * OBS_DIM);
      int4* dst = (int4*)lds_w1[0];
      for (int i = threadIdx.x; i < (HIDDEN * OBS_DIM) / 8; i += WAVES * 32)
        dst[i] = src[i];
    }
    if (s + 1 < NUM_SKILLS) {
      const f16* nxt = W1T + (size_t)(s + 1) * HIDDEN * OBS_DIM;
      for (int i = threadIdx.x; i < 512; i += WAVES * 32)
        __builtin_prefetch(nxt + (size_t)i * 64, 0, 1);
    }
    __syncthreads();
    const f16* w1buf = lds_w1[0];
#endif

    // layer 1: h = relu(obs @ W1[s] + b1[s])
#pragma unroll 1
    for (int nt = 0; nt < HIDDEN / 16; ++nt) {
      v8f acc[TILES];
#pragma unroll
      for (int t = 0; t < TILES; ++t) acc[t] = (v8f){};
#pragma unroll
      for (int kk = 0; kk < OBS_DIM / 32; ++kk) {
        v16h b = load_b(w1buf, OBS_DIM, nt * 16, kk * 32, lane);
#pragma unroll
        for (int t = 0; t < TILES; ++t) {
          v16h a = load_a_gf32(obs, OBS_DIM, rowbase + t * 16, kk * 32, lane);
          acc[t] = wmma_f16(a, b, acc[t]);
        }
      }
      float bias = b1[s * HIDDEN + nt * 16 + n16];
#pragma unroll
      for (int t = 0; t < TILES; ++t)
#pragma unroll
        for (int v = 0; v < 8; ++v) {
          float x = acc[t][v] + bias;
          x = x > 0.f ? x : 0.f;
          lds_h[w][(t * 16 + v + 8 * half) * HIDDEN + nt * 16 + n16] = (f16)x;
        }
    }

    // per-row alphas for this skill (D-fragment row M = v + 8*half)
    float av[TILES][8];
#pragma unroll
    for (int t = 0; t < TILES; ++t)
#pragma unroll
      for (int v = 0; v < 8; ++v)
        av[t][v] = lds_al[w][(t * 16 + v + 8 * half) * 16 + s];

    // layer 2: out = h @ W2[s] + b2[s]; accumulate alpha-weighted mix
    const f16* w2t = W2T + (size_t)s * ACTION_DIM * HIDDEN;
#pragma unroll
    for (int nt = 0; nt < ACTION_DIM / 16; ++nt) {
      v8f acc[TILES];
#pragma unroll
      for (int t = 0; t < TILES; ++t) acc[t] = (v8f){};
#pragma unroll
      for (int kk = 0; kk < HIDDEN / 32; ++kk) {
        v16h b = load_b(w2t, HIDDEN, nt * 16, kk * 32, lane);
#pragma unroll
        for (int t = 0; t < TILES; ++t) {
          v16h a = load_a_lds(lds_h[w] + t * 16 * HIDDEN, HIDDEN, kk * 32, lane);
          acc[t] = wmma_f16(a, b, acc[t]);
        }
      }
      float bias = b2[s * ACTION_DIM + nt * 16 + n16];
#pragma unroll
      for (int t = 0; t < TILES; ++t)
#pragma unroll
        for (int v = 0; v < 8; ++v)
          mix[t][nt][v] += (acc[t][v] + bias) * av[t][v];
    }

#if HAVE_TDM
    // Block done reading buf[s&1]; start DMA of skill s+2 into it.
    __syncthreads();
    if (w == 0 && s + 2 < NUM_SKILLS)
      tdm_load_64k(W1T + (size_t)(s + 2) * HIDDEN * OBS_DIM, lds_w1[s & 1]);
#endif
  }

  // ---- store mixed actions [B,32] ----
#pragma unroll
  for (int t = 0; t < TILES; ++t)
#pragma unroll
    for (int v = 0; v < 8; ++v) {
      size_t row = (size_t)rowbase + t * 16 + v + 8 * half;
      out_mixed[row * ACTION_DIM + n16]      = mix[t][0][v];
      out_mixed[row * ACTION_DIM + 16 + n16] = mix[t][1][v];
    }
}

// ---------------------------------------------------------------------------
extern "C" void kernel_launch(void* const* d_in, const int* in_sizes, int n_in,
                              void* d_out, int out_size, void* d_ws, size_t ws_size,
                              hipStream_t stream) {
  const float* obs   = (const float*)d_in[0];
  const float* goals = (const float*)d_in[1];
  const float* Wa1   = (const float*)d_in[2];
  const float* ba1   = (const float*)d_in[3];
  const float* Wa2   = (const float*)d_in[4];
  const float* ba2   = (const float*)d_in[5];
  const float* W1    = (const float*)d_in[6];
  const float* b1    = (const float*)d_in[7];
  const float* W2    = (const float*)d_in[8];
  const float* b2    = (const float*)d_in[9];

  f16* ws = (f16*)d_ws;
  float* out_mixed  = (float*)d_out;
  float* out_alphas = out_mixed + (size_t)BATCH * ACTION_DIM;

  const int cvt_threads = 256;
  const int cvt_blocks = (WS_F16_TOTAL + cvt_threads - 1) / cvt_threads;
  convert_weights_kernel<<<cvt_blocks, cvt_threads, 0, stream>>>(Wa1, Wa2, W1, W2, ws);

  mop_kernel<<<BATCH / ROWS_PER_BLOCK, WAVES * 32, 0, stream>>>(
      obs, goals, ba1, ba2, b1, b2, ws, out_mixed, out_alphas);
}
